// AFNO2D_35493609734879
// MI455X (gfx1250) — matrix operand order repaired and checked by reading, not compiled, not run on hardware
//
#include <hip/hip_runtime.h>

typedef __attribute__((ext_vector_type(16))) _Float16 v16h;
typedef __attribute__((ext_vector_type(8)))  _Float16 v8h;
typedef __attribute__((ext_vector_type(8)))  float    v8f;

#define NB     8
#define BS     96
#define KDIM   192           // stacked [re(96) | im(96)]
#define HH     256
#define WW     256
#define CC     768
#define WMOD   129           // W/2 + 1
#define NMODES (WMOD * HH)   // 33024
#define LDS_STR 200          // padded LDS row stride (halves) for layer-1 tile
#define SHRINK 0.01f

// ---------------------------------------------------------------------------
// 256-point radix-2 Stockham FFT in LDS. buf = 512 float2 (ping at 0, pong at
// 256). 128 threads per transform (t = 0..127). Natural order in/out, no bit
// reversal. SIGN = -1 forward, +1 inverse (unnormalized).
// ---------------------------------------------------------------------------
template <int SIGN>
__device__ inline void fft256(float2* buf, int t) {
  float2* src = buf;
  float2* dst = buf + 256;
#pragma unroll
  for (int s = 0; s < 8; ++s) {
    const int p = 1 << s;
    const int k = t & (p - 1);
    const int j = ((t - k) << 1) + k;
    const float ang = (float)SIGN * 3.14159265358979323846f * (float)k / (float)p;
    float sn, cs;
    __sincosf(ang, &sn, &cs);
    const float2 a = src[t];
    const float2 b = src[t + 128];
    const float wbx = cs * b.x - sn * b.y;
    const float wby = cs * b.y + sn * b.x;
    dst[j]     = make_float2(a.x + wbx, a.y + wby);
    dst[j + p] = make_float2(a.x - wbx, a.y - wby);
    __syncthreads();
    float2* tmp = src; src = dst; dst = tmp;
  }
}

// Async global->LDS staging helpers (CDNA5 ASYNCcnt path, VGLOBAL encoding).
__device__ inline void async_lds_b32(void* lds, const void* g) {
  unsigned l = (unsigned)(uintptr_t)lds;
  unsigned long long ga = (unsigned long long)(uintptr_t)g;
  asm volatile("global_load_async_to_lds_b32 %0, %1, off" :: "v"(l), "v"(ga) : "memory");
}
__device__ inline void async_lds_b64(void* lds, const void* g) {
  unsigned l = (unsigned)(uintptr_t)lds;
  unsigned long long ga = (unsigned long long)(uintptr_t)g;
  asm volatile("global_load_async_to_lds_b64 %0, %1, off" :: "v"(l), "v"(ga) : "memory");
}
__device__ inline void async_wait0() {
  asm volatile("s_wait_asynccnt 0x0" ::: "memory");
}

// ---------------------------------------------------------------------------
// Prep: expand complex weights into real 192x192 fp16 matrices, stored
// column-major  WT[k][n][r]  so a WMMA B-fragment is one contiguous 16-half
// (32B) load per lane. Also expand biases to [k][192] fp32.
// M = [[Wr, Wi], [-Wi, Wr]]  (rows r = stacked input, cols n = stacked output)
// ---------------------------------------------------------------------------
__global__ void afno_prep(const float* __restrict__ w1, const float* __restrict__ b1,
                          const float* __restrict__ w2, const float* __restrict__ b2,
                          _Float16* __restrict__ WT1, _Float16* __restrict__ WT2,
                          float* __restrict__ BB1, float* __restrict__ BB2) {
  const int idx = blockIdx.x * blockDim.x + threadIdx.x;
  const int total = NB * KDIM * KDIM;
  if (idx < total) {
    const int k   = idx / (KDIM * KDIM);
    const int rem = idx % (KDIM * KDIM);
    const int n = rem / KDIM;      // output (col)
    const int r = rem % KDIM;      // input  (row)
    const int so = n / BS, o = n % BS;
    const int si = r / BS, i = r % BS;
    const size_t wb = ((size_t)(k * BS + i) * BS + o) * 2;
    const float w1r = w1[wb], w1i = w1[wb + 1];
    const float w2r = w2[wb], w2i = w2[wb + 1];
    const float v1 = (si == 0) ? ((so == 0) ? w1r : w1i) : ((so == 0) ? -w1i : w1r);
    const float v2 = (si == 0) ? ((so == 0) ? w2r : w2i) : ((so == 0) ? -w2i : w2r);
    const size_t off = (size_t)k * KDIM * KDIM + (size_t)n * KDIM + r;
    WT1[off] = (_Float16)v1;
    WT2[off] = (_Float16)v2;
  }
  if (idx < NB * KDIM) {
    const int k = idx / KDIM, n = idx % KDIM;
    const int so = n / BS, o = n % BS;
    BB1[idx] = b1[(size_t)(k * BS + o) * 2 + so];
    BB2[idx] = b2[(size_t)(k * BS + o) * 2 + so];
  }
}

// ---------------------------------------------------------------------------
// K1: forward rfft along W. Two real channels packed into one complex FFT.
// Input staged straight into LDS via async loads. bufA: [wm][h][c] float2
// ---------------------------------------------------------------------------
__global__ __launch_bounds__(256) void afno_fftw_fwd(const float* __restrict__ x,
                                                     float2* __restrict__ bufA) {
  __shared__ float2 lds[2][512];
  const int tid = threadIdx.x, sub = tid >> 7, t = tid & 127;
  const int h = blockIdx.y;
  const int cp = blockIdx.x * 2 + sub;
  const int c0 = cp * 2, c1 = c0 + 1;
  float2* buf = lds[sub];
  for (int w = t; w < 256; w += 128) {
    const size_t base = ((size_t)(h * WW + w)) * CC;
    async_lds_b32(&buf[w].x, x + base + c0);
    async_lds_b32(&buf[w].y, x + base + c1);
  }
  async_wait0();
  __syncthreads();
  fft256<-1>(buf, t);
  // unpack two real spectra: X0[k] = (Z[k]+conj(Z[N-k]))/2, X1 = -i(Z[k]-conj(Z[N-k]))/2
  for (int k = t; k <= 128; k += 128) {   // t==0 also covers k=128
    const float2 Z  = buf[k];
    const float2 Zn = buf[(256 - k) & 255];
    const float2 X0 = make_float2(0.5f * (Z.x + Zn.x), 0.5f * (Z.y - Zn.y));
    const float2 X1 = make_float2(0.5f * (Z.y + Zn.y), 0.5f * (Zn.x - Z.x));
    const size_t o = ((size_t)k * HH + h) * CC;
    bufA[o + c0] = X0;
    bufA[o + c1] = X1;
  }
}

// ---------------------------------------------------------------------------
// K2: forward complex FFT along H. Emits fp16 in MLP layout with ortho scale:
// Xh[m*1536 + kb*192 + s*96 + i],  m = wm*256 + h_freq,  c = kb*96 + i
// ---------------------------------------------------------------------------
__global__ __launch_bounds__(256) void afno_ffth_fwd(const float2* __restrict__ bufA,
                                                     _Float16* __restrict__ Xh) {
  __shared__ float2 lds[2][512];
  const int tid = threadIdx.x, sub = tid >> 7, t = tid & 127;
  const int tr = blockIdx.x * 2 + sub;
  const int wm = tr / CC, c = tr % CC;
  float2* buf = lds[sub];
  for (int hh = t; hh < 256; hh += 128)
    async_lds_b64(&buf[hh], &bufA[((size_t)wm * HH + hh) * CC + c]);
  async_wait0();
  __syncthreads();
  fft256<-1>(buf, t);
  const int kb = c / BS, ci = c % BS;
  const float sc = 1.0f / 256.0f;   // rfft2 'ortho' = unnorm / sqrt(H*W)
  for (int hh = t; hh < 256; hh += 128) {
    const float2 v = buf[hh];
    const size_t m = (size_t)wm * HH + hh;
    const size_t o = m * (2 * CC) + (size_t)kb * KDIM + ci;
    Xh[o]      = (_Float16)(v.x * sc);
    Xh[o + BS] = (_Float16)(v.y * sc);
  }
}

// ---------------------------------------------------------------------------
// K3: per-mode block-diagonal complex MLP via WMMA.
// Grid: (NMODES/64, NB). Block = 128 threads = 4 waves.
// Workgroup owns 64 modes (4 mode-tiles) of block kb; wave `w` owns output
// columns [48w, 48w+48) (3 n-tiles) for ALL 4 mode-tiles, so every loaded
// A/B fragment feeds 3-4 back-to-back WMMAs.  acc = 4x3 v8f = 96 VGPRs.
// Layer1 -> bias+relu -> shared LDS tile (A-fragment re-stripe) -> barrier ->
// Layer2 -> bias+softshrink -> fp32 out.  Y layout: [(m*768 + c)*2 + s]
// ---------------------------------------------------------------------------
__global__ __launch_bounds__(128) void afno_mlp(const _Float16* __restrict__ Xh,
                                                const _Float16* __restrict__ WT1,
                                                const _Float16* __restrict__ WT2,
                                                const float* __restrict__ BB1,
                                                const float* __restrict__ BB2,
                                                float* __restrict__ Y) {
  __shared__ _Float16 o1s[64 * LDS_STR];   // 64 modes x 200 halves = 25.6 KB
  const int lane = threadIdx.x & 31;
  const int wave = threadIdx.x >> 5;
  const int half = lane >> 4;
  const int l16  = lane & 15;
  const int kb   = blockIdx.y;
  const size_t mode0 = (size_t)blockIdx.x * 64;
  const int ncol0 = wave * 48;             // this wave's N-slice

  const _Float16* Xbase = Xh + mode0 * (2 * CC) + (size_t)kb * KDIM;
  const _Float16* W1    = WT1 + (size_t)kb * KDIM * KDIM;
  const _Float16* W2    = WT2 + (size_t)kb * KDIM * KDIM;
  __builtin_prefetch(W2 + (size_t)(ncol0 + l16) * KDIM, 0, 1);

  union Frag { v16h v; v8h h[2]; };

  // ---------------- layer 1 ----------------
  v8f acc[4][3];
#pragma unroll
  for (int mt = 0; mt < 4; ++mt)
#pragma unroll
    for (int nt = 0; nt < 3; ++nt) acc[mt][nt] = (v8f)0.0f;

  for (int kk0 = 0; kk0 < KDIM; kk0 += 32) {
    Frag a[4];
#pragma unroll
    for (int mt = 0; mt < 4; ++mt) {
      const _Float16* ap = Xbase + (size_t)(mt * 16 + l16) * (2 * CC) + kk0 + half * 16;
      a[mt].h[0] = *(const v8h*)(ap);
      a[mt].h[1] = *(const v8h*)(ap + 8);
    }
#pragma unroll
    for (int nt = 0; nt < 3; ++nt) {
      const v16h b = *(const v16h*)(W1 + (size_t)(ncol0 + nt * 16 + l16) * KDIM + kk0 + half * 16);
#pragma unroll
      for (int mt = 0; mt < 4; ++mt)
        acc[mt][nt] = __builtin_amdgcn_wmma_f32_16x16x32_f16(false, a[mt].v, false, b,
                                                             (short)0, acc[mt][nt], false, false);
    }
  }

  // bias + relu -> shared LDS tile, re-striped to A-fragment layout
#pragma unroll
  for (int nt = 0; nt < 3; ++nt) {
    const int n = ncol0 + nt * 16 + l16;
    const float bias = BB1[kb * KDIM + n];
#pragma unroll
    for (int mt = 0; mt < 4; ++mt)
#pragma unroll
      for (int j = 0; j < 8; ++j) {
        const int m = mt * 16 + j + half * 8;     // D-fragment row mapping
        float v = acc[mt][nt][j] + bias;
        v = v > 0.0f ? v : 0.0f;
        o1s[m * LDS_STR + n] = (_Float16)v;
      }
  }
  __syncthreads();

  // ---------------- layer 2 ----------------
  v8f acc2[4][3];
#pragma unroll
  for (int mt = 0; mt < 4; ++mt)
#pragma unroll
    for (int nt = 0; nt < 3; ++nt) acc2[mt][nt] = (v8f)0.0f;

  for (int kk0 = 0; kk0 < KDIM; kk0 += 32) {
    Frag a[4];
#pragma unroll
    for (int mt = 0; mt < 4; ++mt) {
      const _Float16* ap = o1s + (size_t)(mt * 16 + l16) * LDS_STR + kk0 + half * 16;
      a[mt].h[0] = *(const v8h*)(ap);
      a[mt].h[1] = *(const v8h*)(ap + 8);
    }
#pragma unroll
    for (int nt = 0; nt < 3; ++nt) {
      const v16h b = *(const v16h*)(W2 + (size_t)(ncol0 + nt * 16 + l16) * KDIM + kk0 + half * 16);
#pragma unroll
      for (int mt = 0; mt < 4; ++mt)
        acc2[mt][nt] = __builtin_amdgcn_wmma_f32_16x16x32_f16(false, a[mt].v, false, b,
                                                              (short)0, acc2[mt][nt], false, false);
    }
  }

  // bias + softshrink -> fp32
#pragma unroll
  for (int nt = 0; nt < 3; ++nt) {
    const int n = ncol0 + nt * 16 + l16;
    const int s = (n >= BS) ? 1 : 0;
    const int i = n - s * BS;
    const int c = kb * BS + i;
    const float bias = BB2[kb * KDIM + n];
#pragma unroll
    for (int mt = 0; mt < 4; ++mt)
#pragma unroll
      for (int j = 0; j < 8; ++j) {
        const size_t m = mode0 + mt * 16 + j + half * 8;
        float v = acc2[mt][nt][j] + bias;
        v = (v > SHRINK) ? (v - SHRINK) : ((v < -SHRINK) ? (v + SHRINK) : 0.0f);
        Y[(m * CC + c) * 2 + s] = v;
      }
  }
}

// ---------------------------------------------------------------------------
// K4: inverse complex FFT along H.  bufD layout: [h][c][wm] float2
// ---------------------------------------------------------------------------
__global__ __launch_bounds__(256) void afno_ffth_inv(const float* __restrict__ Yc,
                                                     float2* __restrict__ bufD) {
  __shared__ float2 lds[2][512];
  const int tid = threadIdx.x, sub = tid >> 7, t = tid & 127;
  const int tr = blockIdx.x * 2 + sub;
  const int wm = tr / CC, c = tr % CC;
  float2* buf = lds[sub];
  for (int hh = t; hh < 256; hh += 128) {
    const size_t m = (size_t)wm * HH + hh;
    async_lds_b64(&buf[hh], Yc + (m * CC + c) * 2);
  }
  async_wait0();
  __syncthreads();
  fft256<1>(buf, t);
  for (int hh = t; hh < 256; hh += 128)
    bufD[((size_t)hh * CC + c) * WMOD + wm] = buf[hh];
}

// ---------------------------------------------------------------------------
// K5: inverse rfft along W (Hermitian extension), two channels per complex
// transform, + residual. out = Re/Im(ifft(Z))/256 + x
// ---------------------------------------------------------------------------
__global__ __launch_bounds__(256) void afno_fftw_inv(const float2* __restrict__ bufD,
                                                     const float* __restrict__ x,
                                                     float* __restrict__ out) {
  __shared__ float2 lds[2][512];
  const int tid = threadIdx.x, sub = tid >> 7, t = tid & 127;
  const int h = blockIdx.y;
  const int cp = blockIdx.x * 2 + sub;
  const int c0 = cp * 2, c1 = c0 + 1;
  float2* buf = lds[sub];
  {
    const int k = t;
    const float2 X0 = bufD[((size_t)h * CC + c0) * WMOD + k];
    const float2 X1 = bufD[((size_t)h * CC + c1) * WMOD + k];
    buf[k] = make_float2(X0.x - X1.y, X0.y + X1.x);            // Z[k] = X0 + i*X1
    if (k > 0)                                                 // Z[N-k] = conj(X0)+i*conj(X1)
      buf[256 - k] = make_float2(X0.x + X1.y, X1.x - X0.y);
    if (t == 0) {
      const float2 Y0 = bufD[((size_t)h * CC + c0) * WMOD + 128];
      const float2 Y1 = bufD[((size_t)h * CC + c1) * WMOD + 128];
      buf[128] = make_float2(Y0.x - Y1.y, Y0.y + Y1.x);
    }
  }
  __syncthreads();
  fft256<1>(buf, t);
  const float sc = 1.0f / 256.0f;   // irfft2 'ortho'
  for (int w = t; w < 256; w += 128) {
    const size_t base = ((size_t)(h * WW + w)) * CC;
    out[base + c0] = buf[w].x * sc + x[base + c0];
    out[base + c1] = buf[w].y * sc + x[base + c1];
  }
}

// ---------------------------------------------------------------------------
extern "C" void kernel_launch(void* const* d_in, const int* in_sizes, int n_in,
                              void* d_out, int out_size, void* d_ws, size_t ws_size,
                              hipStream_t stream) {
  const float* x  = (const float*)d_in[0];
  const float* w1 = (const float*)d_in[1];
  const float* b1 = (const float*)d_in[2];
  const float* w2 = (const float*)d_in[3];
  const float* b2 = (const float*)d_in[4];
  float* out = (float*)d_out;

  char* ws = (char*)d_ws;
  const size_t R0_BYTES = (size_t)WMOD * HH * CC * 2 * sizeof(float); // 202,899,456
  // Region 0: bufA (fwd) -> reused as Y (MLP output)
  float2* bufA = (float2*)ws;
  float*  Yc   = (float*)ws;
  // Region 1: Xh fp16 (fwd staging) -> reused as bufD (inverse staging)
  _Float16* Xh   = (_Float16*)(ws + R0_BYTES);
  float2*   bufD = (float2*)(ws + R0_BYTES);
  // Weights / biases
  char* wp = ws + 2 * R0_BYTES;
  _Float16* WT1 = (_Float16*)wp; wp += (size_t)NB * KDIM * KDIM * sizeof(_Float16);
  _Float16* WT2 = (_Float16*)wp; wp += (size_t)NB * KDIM * KDIM * sizeof(_Float16);
  float* BB1 = (float*)wp;       wp += (size_t)NB * KDIM * sizeof(float);
  float* BB2 = (float*)wp;

  // 0) expand weights (1.2 MB, L2-resident thereafter)
  afno_prep<<<dim3((NB * KDIM * KDIM + 255) / 256), dim3(256), 0, stream>>>(
      w1, b1, w2, b2, WT1, WT2, BB1, BB2);

  // 1) forward rfft along W (channel-pair packed): 384 pairs, 2 per block
  afno_fftw_fwd<<<dim3(CC / 4, HH), dim3(256), 0, stream>>>(x, bufA);

  // 2) forward FFT along H -> fp16 MLP layout (129*768 transforms, 2 per block)
  afno_ffth_fwd<<<dim3(WMOD * CC / 2), dim3(256), 0, stream>>>(bufA, Xh);

  // 3) WMMA block-diagonal complex MLP: 64 modes x 1 block per workgroup
  afno_mlp<<<dim3(NMODES / 64, NB), dim3(128), 0, stream>>>(Xh, WT1, WT2, BB1, BB2, Yc);

  // 4) inverse FFT along H
  afno_ffth_inv<<<dim3(WMOD * CC / 2), dim3(256), 0, stream>>>(Yc, bufD);

  // 5) inverse rfft along W + residual
  afno_fftw_inv<<<dim3(CC / 4, HH), dim3(256), 0, stream>>>(bufD, x, out);
}
